// SDPACoreAttention_386547057127
// MI455X (gfx1250) — matrix operand order repaired and checked by reading, not compiled, not run on hardware
//
#include <hip/hip_runtime.h>

#define T_LOCAL 2048
#define NQH 16
#define NKH 8
#define HN 128
#define SGLB 4096
#define WIN 1024
#define REP (NQH / NKH)
#define BLK 64

typedef __attribute__((ext_vector_type(16))) __bf16 v16bf;
typedef __attribute__((ext_vector_type(8)))  float  v8f;

union BF16x16 { v16bf v; unsigned u[8]; uint4 q[2]; };

template<bool B> struct BoolC { static constexpr bool value = B; };

// allow plain ALU to cross the fence, pin WMMA/VMEM/DS (caps operand prefetch)
#define SCHED_FENCE() __builtin_amdgcn_sched_barrier(0x407)

// pack two f32 -> two bf16 (truncation) in ONE v_perm_b32
__device__ __forceinline__ unsigned pack_bf(float lo, float hi) {
  return __builtin_amdgcn_perm(__float_as_uint(hi), __float_as_uint(lo), 0x07060302u);
}

// gathered zigzag layout: global position g -> offset in gathered K/V (CP=2, S=4096)
__device__ __forceinline__ int g2l(int g) {
  return g < 1024 ? g : (g < 3072 ? g + 1024 : g - 2048);
}

// LDS row strides in u32 units (row starts 16B aligned, strides spread banks)
#define KSTR32 68   // K block [64 keys][128 d]   : byte stride 272
#define VSTR32 36   // V^T     [128 d][64 keys]   : byte stride 144
#define PSTR32 36   // P tile  [16 rows][64 keys] : byte stride 144

__global__ __launch_bounds__(64)
void sdpa_zigzag_swa_kernel(const float* __restrict__ Q,
                            const float* __restrict__ K,
                            const float* __restrict__ V,
                            float* __restrict__ Out)
{
  __shared__ unsigned kb32[BLK * KSTR32];       // K block, bf16 pairs
  __shared__ unsigned vb32[HN * VSTR32];        // V^T block, bf16 key-pairs
  __shared__ unsigned pl32[2 * 16 * PSTR32];    // per-wave P tiles

  const int tid  = threadIdx.x;
  const int lane = tid & 31;
  const int wave = tid >> 5;
  const int nn   = lane & 15;   // N index within 16-wide tile (C/D & B layouts)
  const int seg  = lane >> 4;   // half-wave segment

  const int qt = blockIdx.x;                 // 0..127 q-tile
  const int hk = blockIdx.y;                 // 0..7 kv head
  const int h  = hk * REP + wave;            // q head

  const int q0     = qt * 16;
  const int r_base = (q0 < T_LOCAL / 2) ? q0 : q0 + (SGLB - T_LOCAL);

  // softmax in exp2 domain: fold log2(e) into the score scale
  const float scale2 = 0.08838834764831844f * 1.4426950408889634f;
  const float NEGF   = -1e30f;

  // ---- Q tile -> WMMA A-operand layout (bf16), 4 chunks of K=32
  BF16x16 qa[4];
  {
    const float* qrow = Q + ((size_t)(q0 + nn) * NQH + h) * HN;
#pragma unroll
    for (int c = 0; c < 4; ++c) {
      const int b0 = c * 32 + seg * 8;
#pragma unroll
      for (int k = 0; k < 4; ++k) {
        qa[c].u[k]     = pack_bf(qrow[b0 + 2 * k],      qrow[b0 + 2 * k + 1]);
        qa[c].u[4 + k] = pack_bf(qrow[b0 + 16 + 2 * k], qrow[b0 + 16 + 2 * k + 1]);
      }
    }
  }

  const v8f zero8 = {0.f, 0.f, 0.f, 0.f, 0.f, 0.f, 0.f, 0.f};
  v8f o[8];
#pragma unroll
  for (int n = 0; n < 8; ++n) o[n] = zero8;
  float mrow[8], lrow[8];
#pragma unroll
  for (int i = 0; i < 8; ++i) { mrow[i] = NEGF; lrow[i] = 0.f; }

  int gb0 = r_base - (WIN - 1);
  if (gb0 < 0) gb0 = 0;
  gb0 &= ~(BLK - 1);
  const int g_hi = r_base + 15;

  for (int gb = gb0; gb <= g_hi; gb += BLK) {
    __syncthreads();  // previous iteration's kb/vb reads complete

    // ---- cooperative stage: K block [64 keys][128 d] as bf16 pairs (once per WG)
    {
      const int key = tid;  // 64 threads <-> 64 keys
      int g = gb + key; if (g > SGLB - 1) g = SGLB - 1;
      const float* kp = K + ((size_t)g2l(g) * NKH + hk) * HN;
      unsigned* dst = &kb32[key * KSTR32];
#pragma unroll 8
      for (int j = 0; j < 64; ++j)
        dst[j] = pack_bf(kp[2 * j], kp[2 * j + 1]);
    }
    // ---- cooperative stage: V block transposed [128 d][64 keys], bf16 key-pairs
    {
      const int kp2 = tid & 31;        // key pair index 0..31
      const int db  = (tid >> 5) * 64; // 64-dim slab
      int ga = gb + 2 * kp2;     if (ga > SGLB - 1) ga = SGLB - 1;
      int gc = gb + 2 * kp2 + 1; if (gc > SGLB - 1) gc = SGLB - 1;
      const float* va = V + ((size_t)g2l(ga) * NKH + hk) * HN + db;
      const float* vc = V + ((size_t)g2l(gc) * NKH + hk) * HN + db;
#pragma unroll 8
      for (int j = 0; j < 64; ++j)
        vb32[(db + j) * VSTR32 + kp2] = pack_bf(va[j], vc[j]);
    }
    __syncthreads();  // kb/vb ready

    // ---- prefetch next block's K/V cachelines into cache during compute
    {
      const int gnext = gb + BLK;
      if (gnext <= g_hi) {  // wave-uniform
        int gk = gnext + tid; if (gk > SGLB - 1) gk = SGLB - 1;
        const float* kpn = K + ((size_t)g2l(gk) * NKH + hk) * HN;
        __builtin_prefetch(kpn,      0, 3);
        __builtin_prefetch(kpn + 32, 0, 3);
        __builtin_prefetch(kpn + 64, 0, 3);
        __builtin_prefetch(kpn + 96, 0, 3);
        const int kp2 = tid & 31;
        const int db  = (tid >> 5) * 64;
        int ga = gnext + 2 * kp2;     if (ga > SGLB - 1) ga = SGLB - 1;
        int gc = gnext + 2 * kp2 + 1; if (gc > SGLB - 1) gc = SGLB - 1;
        const float* van = V + ((size_t)g2l(ga) * NKH + hk) * HN + db;
        const float* vcn = V + ((size_t)g2l(gc) * NKH + hk) * HN + db;
        __builtin_prefetch(van,      0, 3);
        __builtin_prefetch(van + 32, 0, 3);
        __builtin_prefetch(vcn,      0, 3);
        __builtin_prefetch(vcn + 32, 0, 3);
      }
    }

    // ---- scores: S[16 q x 64 keys] = Q * K^T, four 16-key subtiles, 4 K-chunks
    v8f sc[4];
#pragma unroll
    for (int st = 0; st < 4; ++st) {
      if (st == 2) SCHED_FENCE();  // cap K-operand prefetch
      v8f acc = zero8;
      const int krow = (st * 16 + nn) * KSTR32 + seg * 4;
#pragma unroll
      for (int c = 0; c < 4; ++c) {
        BF16x16 kb;
        const uint4* p = (const uint4*)&kb32[krow + c * 16];
        kb.q[0] = p[0];
        kb.q[1] = p[2];
        acc = __builtin_amdgcn_wmma_f32_16x16x32_bf16(
            false, qa[c].v, false, kb.v, (short)0, acc, false, false);
      }
      sc[st] = acc;
    }

    // ---- online softmax; masked variant only for window-edge / diagonal blocks
    unsigned short* pw = (unsigned short*)&pl32[wave * 16 * PSTR32];
    const bool full = (gb >= r_base - (WIN - 16)) && (gb <= r_base - (BLK - 1));
    v8f al = zero8;  // per-row rescale factors for O

    auto upd = [&](auto mc) {
      constexpr bool MASKED = decltype(mc)::value;
#pragma unroll
      for (int i = 0; i < 8; ++i) {
        const int r = r_base + i + 8 * seg;
        float a0 = sc[0][i] * scale2;
        float a1 = sc[1][i] * scale2;
        float a2 = sc[2][i] * scale2;
        float a3 = sc[3][i] * scale2;
        if constexpr (MASKED) {
          const int c0 = gb + nn, c1 = c0 + 16, c2 = c0 + 32, c3 = c0 + 48;
          a0 = (c0 <= r && c0 + (WIN - 1) >= r) ? a0 : NEGF;
          a1 = (c1 <= r && c1 + (WIN - 1) >= r) ? a1 : NEGF;
          a2 = (c2 <= r && c2 + (WIN - 1) >= r) ? a2 : NEGF;
          a3 = (c3 <= r && c3 + (WIN - 1) >= r) ? a3 : NEGF;
        }
        float mx = fmaxf(fmaxf(a0, a1), fmaxf(a2, a3));
        mx = fmaxf(mx, __shfl_xor(mx, 1, 16));
        mx = fmaxf(mx, __shfl_xor(mx, 2, 16));
        mx = fmaxf(mx, __shfl_xor(mx, 4, 16));
        mx = fmaxf(mx, __shfl_xor(mx, 8, 16));
        const float mn    = fmaxf(mrow[i], mx);
        const float alpha = exp2f(mrow[i] - mn);
        float p0, p1, p2, p3;
        if constexpr (MASKED) {
          p0 = (a0 == NEGF) ? 0.f : exp2f(a0 - mn);
          p1 = (a1 == NEGF) ? 0.f : exp2f(a1 - mn);
          p2 = (a2 == NEGF) ? 0.f : exp2f(a2 - mn);
          p3 = (a3 == NEGF) ? 0.f : exp2f(a3 - mn);
        } else {
          p0 = exp2f(a0 - mn);
          p1 = exp2f(a1 - mn);
          p2 = exp2f(a2 - mn);
          p3 = exp2f(a3 - mn);
        }
        float ps = (p0 + p1) + (p2 + p3);
        ps += __shfl_xor(ps, 1, 16);
        ps += __shfl_xor(ps, 2, 16);
        ps += __shfl_xor(ps, 4, 16);
        ps += __shfl_xor(ps, 8, 16);
        lrow[i] = lrow[i] * alpha + ps;
        mrow[i] = mn;
        al[i]   = alpha;
        const int row = (i + 8 * seg) * (2 * PSTR32);
        pw[row + nn]      = (unsigned short)(__float_as_uint(p0) >> 16);
        pw[row + 16 + nn] = (unsigned short)(__float_as_uint(p1) >> 16);
        pw[row + 32 + nn] = (unsigned short)(__float_as_uint(p2) >> 16);
        pw[row + 48 + nn] = (unsigned short)(__float_as_uint(p3) >> 16);
      }
    };
    if (full) upd(BoolC<false>{});
    else      upd(BoolC<true>{});

    // ---- rescale O once, as a compact vector pass (lowers to v_pk_mul_f32)
#pragma unroll
    for (int n = 0; n < 8; ++n) o[n] *= al;

    // ---- reload P tile (16 x 64) as two WMMA A-operands
    BF16x16 pa0, pa1;
    {
      const unsigned* pb = &pl32[wave * 16 * PSTR32 + nn * PSTR32 + seg * 4];
      pa0.q[0] = *(const uint4*)(pb);       // keys  0..15 half (seg split)
      pa0.q[1] = *(const uint4*)(pb + 8);   // keys 16..31
      pa1.q[0] = *(const uint4*)(pb + 16);  // keys 32..47
      pa1.q[1] = *(const uint4*)(pb + 24);  // keys 48..63
    }

    // ---- PV: O[16 q x 128 d] += P * V; fully unrolled, fenced every 2 tiles
#pragma unroll
    for (int n = 0; n < 8; ++n) {
      if (n == 2 || n == 4 || n == 6) SCHED_FENCE();
      const unsigned* vp = &vb32[(n * 16 + nn) * VSTR32 + seg * 4];
      BF16x16 vlo, vhi;
      vlo.q[0] = *(const uint4*)(vp);
      vlo.q[1] = *(const uint4*)(vp + 8);
      vhi.q[0] = *(const uint4*)(vp + 16);
      vhi.q[1] = *(const uint4*)(vp + 24);
      o[n] = __builtin_amdgcn_wmma_f32_16x16x32_bf16(
          false, pa0.v, false, vlo.v, (short)0, o[n], false, false);
      o[n] = __builtin_amdgcn_wmma_f32_16x16x32_bf16(
          false, pa1.v, false, vhi.v, (short)0, o[n], false, false);
    }
  }

  // ---- normalize and store fp32 output [t_local, NQ*HN]
#pragma unroll
  for (int i = 0; i < 8; ++i) {
    const int   t   = q0 + i + 8 * seg;
    const float inv = 1.f / lrow[i];
#pragma unroll
    for (int n = 0; n < 8; ++n)
      Out[(size_t)t * (NQH * HN) + h * HN + n * 16 + nn] = o[n][i] * inv;
  }
}

extern "C" void kernel_launch(void* const* d_in, const int* in_sizes, int n_in,
                              void* d_out, int out_size, void* d_ws, size_t ws_size,
                              hipStream_t stream) {
  (void)in_sizes; (void)n_in; (void)out_size; (void)d_ws; (void)ws_size;
  const float* Q = (const float*)d_in[0];   // [2048, 16, 128]
  const float* K = (const float*)d_in[1];   // [4096, 8, 128] gathered zigzag
  const float* V = (const float*)d_in[2];   // [4096, 8, 128]
  float* Out = (float*)d_out;               // [2048, 2048]
  dim3 grid(T_LOCAL / 16, NKH);
  sdpa_zigzag_swa_kernel<<<grid, 64, 0, stream>>>(Q, K, V, Out);
}
